// MixtureOfMambaBlock_39101382262867
// MI455X (gfx1250) — matrix operand order repaired and checked by reading, not compiled, not run on hardware
//
#include <hip/hip_runtime.h>

// ---------------------------------------------------------------------------
// MixtureOfMamba block for gfx1250 (MI455X), bf16 WMMA GEMM core.
// All matrix math runs through v_wmma_f32_16x16x32_bf16.
// ---------------------------------------------------------------------------

typedef __bf16 bf16;
typedef __attribute__((ext_vector_type(16))) __bf16 v16bf;
typedef __attribute__((ext_vector_type(8)))  float  v8f;
typedef __attribute__((ext_vector_type(4)))  int    v4i;

// CDNA5 async global->LDS copy (ASYNCcnt-tracked), guarded for portability.
#if defined(__has_builtin)
#  if __has_builtin(__builtin_amdgcn_global_load_async_to_lds_b128) && \
      __has_builtin(__builtin_amdgcn_s_wait_asynccnt)
#    define HAVE_ASYNC_LDS 1
#  endif
#endif
#ifndef HAVE_ASYNC_LDS
#  define HAVE_ASYNC_LDS 0
#endif

#if HAVE_ASYNC_LDS
// Builtin signature (from hipcc diagnostic): param0 is addrspace(1) int4*,
// i.e. 'int __vector(4) __device__ *'. Cast helpers:
__device__ static inline __attribute__((address_space(1))) v4i*
to_global_v4i(const void* p)
{
    return (__attribute__((address_space(1))) v4i*)(v4i*)const_cast<void*>(p);
}
__device__ static inline __attribute__((address_space(3))) v4i*
to_lds_v4i(void* p)
{
    return (__attribute__((address_space(3))) v4i*)(v4i*)p;
}
#endif

// Problem constants (match reference)
constexpr int Bc     = 2;
constexpr int Tc     = 2048;
constexpr int Dc     = 1024;
constexpr int Sc     = 64;     // D_STATE
constexpr int INNERc = 2048;   // D * EXPAND
constexpr int HIDc   = 4096;   // D * FF_MULT
constexpr int Ec     = 4;
constexpr int Mtok   = Bc * Tc;   // 4096 tokens

// GEMM tiling: 128x64 block tile, 8 waves, each wave owns a 32x32 macro-tile
// (2x2 WMMA tiles -> 4 v_wmma per K-step with A/B fragment reuse).
constexpr int BM  = 128;
constexpr int BN  = 64;
constexpr int BK  = 32;
constexpr int LDA = BK + 24;  // 56 halves: rows 112B (16B-aligned for b128 async,
                              // dword stride 28 -> conflict-free frag reads)
constexpr int LDK = BK + 8;   // 40 halves for transposed-B tile (dword stride 20)

// ---------------------------------------------------------------------------
// WMMA GEMM: C[M,N] = A[M,K](bf16) @ W[K,N](bf16) + bias[N], fp32 out.
// M % 128 == 0, N % 64 == 0, K % 32 == 0 (guaranteed by caller).
// ---------------------------------------------------------------------------
__global__ __launch_bounds__(256)
void gemm_bf16_wmma(const bf16* __restrict__ A, const bf16* __restrict__ Bw,
                    const float* __restrict__ bias, float* __restrict__ C,
                    int Ndim, int Kdim)
{
    __shared__ bf16 sA [BM * LDA];   // 14336 B
    __shared__ bf16 sBt[BN * LDK];   //  5120 B (B tile stored transposed: [n][k])

    const int tid  = threadIdx.x;
    const int lane = tid & 31;
    const int wave = tid >> 5;      // 0..7
    const int wm   = wave & 3;      // M macro-tile 0..3 (rows wm*32)
    const int wn   = wave >> 2;     // N macro-tile 0..1 (cols wn*32)

    const int block_m = blockIdx.y * BM;
    const int block_n = blockIdx.x * BN;

    const int lrow = lane & 15;
    const int akb  = (lane < 16) ? 0 : 16;   // A fragment K base (contiguous k = 16h+s)
    const int bkb  = (lane < 16) ? 0 : 16;   // B fragment K base

    const int am0 = wm * 32 + lrow;          // A fragment rows (two tiles)
    const int am1 = am0 + 16;
    const int bn0 = wn * 32 + lrow;          // B fragment cols (two tiles)
    const int bn1 = bn0 + 16;

    // cooperative-load coordinates
    const int ar = tid >> 1;                 // 0..127 (A row)
    const int ac = (tid & 1) << 4;           // 0 / 16  (A col, 16 halves each)
    const int br = tid >> 3;                 // 0..31  (B row = k)
    const int bc = (tid & 7) << 3;           // 0..56  (B col, 8 halves)

    v8f acc[2][2] = {{{}, {}}, {{}, {}}};

    for (int k0 = 0; k0 < Kdim; k0 += BK) {
        const bf16* gA = A  + (size_t)(block_m + ar) * Kdim + k0 + ac;
        const bf16* gB = Bw + (size_t)(k0 + br) * Ndim + block_n + bc;

        // ---- A tile: 128x32, 16 halves (32B) per thread ----
#if HAVE_ASYNC_LDS
        __builtin_amdgcn_global_load_async_to_lds_b128(
            to_global_v4i(gA), to_lds_v4i(&sA[ar * LDA + ac]), 0, 0);
        __builtin_amdgcn_global_load_async_to_lds_b128(
            to_global_v4i(gA + 8), to_lds_v4i(&sA[ar * LDA + ac + 8]), 0, 0);
#else
        {
            const uint4 v0 = *reinterpret_cast<const uint4*>(gA);
            const uint4 v1 = *reinterpret_cast<const uint4*>(gA + 8);
            unsigned int* dst = reinterpret_cast<unsigned int*>(&sA[ar * LDA + ac]);
            dst[0] = v0.x; dst[1] = v0.y; dst[2] = v0.z; dst[3] = v0.w;
            dst[4] = v1.x; dst[5] = v1.y; dst[6] = v1.z; dst[7] = v1.w;
        }
#endif
        // ---- B tile: 32x64 loaded coalesced, stored transposed [n][k] ----
        {
            union { uint4 v; bf16 h[8]; } u;
            u.v = *reinterpret_cast<const uint4*>(gB);
#pragma unroll
            for (int i = 0; i < 8; i++) sBt[(bc + i) * LDK + br] = u.h[i];
        }

        // prefetch next K tile while this one is being staged
        if (k0 + BK < Kdim) {
            __builtin_prefetch(gA + BK, 0, 0);
            __builtin_prefetch(Bw + (size_t)(k0 + BK + br) * Ndim + block_n + bc, 0, 0);
        }

#if HAVE_ASYNC_LDS
        __builtin_amdgcn_s_wait_asynccnt(0);
#endif
        __syncthreads();

        // ---- fragments: all contiguous dword reads from LDS ----
        union { unsigned int u[8]; v16bf v; } a0, a1, b0, b1;
        {
            const unsigned int* p;
            p = reinterpret_cast<const unsigned int*>(&sA[am0 * LDA + akb]);
#pragma unroll
            for (int j = 0; j < 4; j++) a0.u[j] = p[j];
            p = reinterpret_cast<const unsigned int*>(&sA[am0 * LDA + akb + 8]);
#pragma unroll
            for (int j = 0; j < 4; j++) a0.u[4 + j] = p[j];
            p = reinterpret_cast<const unsigned int*>(&sA[am1 * LDA + akb]);
#pragma unroll
            for (int j = 0; j < 4; j++) a1.u[j] = p[j];
            p = reinterpret_cast<const unsigned int*>(&sA[am1 * LDA + akb + 8]);
#pragma unroll
            for (int j = 0; j < 4; j++) a1.u[4 + j] = p[j];
            p = reinterpret_cast<const unsigned int*>(&sBt[bn0 * LDK + bkb]);
#pragma unroll
            for (int j = 0; j < 8; j++) b0.u[j] = p[j];
            p = reinterpret_cast<const unsigned int*>(&sBt[bn1 * LDK + bkb]);
#pragma unroll
            for (int j = 0; j < 8; j++) b1.u[j] = p[j];
        }

        acc[0][0] = __builtin_amdgcn_wmma_f32_16x16x32_bf16(
            false, a0.v, false, b0.v, (short)0, acc[0][0], false, false);
        acc[0][1] = __builtin_amdgcn_wmma_f32_16x16x32_bf16(
            false, a0.v, false, b1.v, (short)0, acc[0][1], false, false);
        acc[1][0] = __builtin_amdgcn_wmma_f32_16x16x32_bf16(
            false, a1.v, false, b0.v, (short)0, acc[1][0], false, false);
        acc[1][1] = __builtin_amdgcn_wmma_f32_16x16x32_bf16(
            false, a1.v, false, b1.v, (short)0, acc[1][1], false, false);
        __syncthreads();
    }

    // ---- epilogue: C/D layout: elem r -> M = r + (lane>=16 ? 8 : 0) ----
    const int msel = (lane < 16) ? 0 : 8;
#pragma unroll
    for (int ti = 0; ti < 2; ti++) {
        const int mbase = block_m + wm * 32 + ti * 16 + msel;
#pragma unroll
        for (int tj = 0; tj < 2; tj++) {
            const int nc = block_n + wn * 32 + tj * 16 + lrow;
            const float bv = bias[nc];
#pragma unroll
            for (int r = 0; r < 8; r++)
                C[(size_t)(mbase + r) * Ndim + nc] = acc[ti][tj][r] + bv;
        }
    }
}

// ---------------------------------------------------------------------------
// Elementwise / reduction kernels
// ---------------------------------------------------------------------------

__global__ void f32_to_bf16(const float* __restrict__ in, bf16* __restrict__ out,
                            size_t n)
{
    size_t i = (size_t)blockIdx.x * blockDim.x + threadIdx.x;
    const size_t stride = (size_t)gridDim.x * blockDim.x;
    for (; i < n; i += stride) out[i] = (bf16)in[i];
}

__global__ __launch_bounds__(256)
void rmsnorm_to_bf16(const float* __restrict__ x, const float* __restrict__ w,
                     bf16* __restrict__ out, int Dd)
{
    const int row = blockIdx.x;
    const float* xr = x + (size_t)row * Dd;
    float ss = 0.f;
    for (int i = threadIdx.x; i < Dd; i += 256) { float v = xr[i]; ss += v * v; }
    for (int o = 16; o > 0; o >>= 1) ss += __shfl_down(ss, o, 32);
    __shared__ float part[8];
    __shared__ float nrs;
    if ((threadIdx.x & 31) == 0) part[threadIdx.x >> 5] = ss;
    __syncthreads();
    if (threadIdx.x == 0) {
        float t = 0.f;
        for (int i = 0; i < 8; i++) t += part[i];
        nrs = rsqrtf(t / (float)Dd + 1e-6f);
    }
    __syncthreads();
    const float nr = nrs;
    bf16* orow = out + (size_t)row * Dd;
    for (int i = threadIdx.x; i < Dd; i += 256)
        orow[i] = (bf16)(xr[i] * nr * w[i]);
}

// causal depthwise conv (K=3) + bias + silu over x_main half of xz
__global__ void conv_silu(const float* __restrict__ xz, const float* __restrict__ cw,
                          const float* __restrict__ cb,
                          float* __restrict__ xm_f, bf16* __restrict__ xm_bf)
{
    const size_t n = (size_t)Mtok * INNERc;
    size_t idx = (size_t)blockIdx.x * blockDim.x + threadIdx.x;
    if (idx >= n) return;
    const int c   = (int)(idx & (INNERc - 1));
    const int row = (int)(idx >> 11);          // INNERc == 2048
    const int t   = row & (Tc - 1);
    const float w0 = cw[c * 3 + 0], w1 = cw[c * 3 + 1], w2 = cw[c * 3 + 2];
    const float x0 = xz[(size_t)row * (2 * INNERc) + c];
    const float x1 = (t >= 1) ? xz[(size_t)(row - 1) * (2 * INNERc) + c] : 0.f;
    const float x2 = (t >= 2) ? xz[(size_t)(row - 2) * (2 * INNERc) + c] : 0.f;
    const float a = w0 * x2 + w1 * x1 + w2 * x0 + cb[c];
    const float s = a / (1.f + __expf(-a));    // silu
    xm_f[idx]  = s;
    xm_bf[idx] = (bf16)s;
}

// sequential SSM scan: 128 independent (batch, state) chains over T
__global__ void ssm_scan(const float* __restrict__ dtlin, const float* __restrict__ Bm,
                         const float* __restrict__ Cm, float* __restrict__ y)
{
    const int tid = threadIdx.x;               // 0..127
    const int b = tid >> 6, s = tid & 63;
    float st = 0.f;
    for (int t = 0; t < Tc; t++) {
        const size_t off = ((size_t)(b * Tc + t)) * Sc + s;
        const float d = 1.f / (1.f + __expf(-dtlin[off]));
        st = (1.f - d) * st + d * Bm[off];
        y[off] = Cm[off] * st;
    }
}

// layernorm over the S=64 state dim, output bf16
__global__ __launch_bounds__(64)
void layernorm64_to_bf16(const float* __restrict__ y, bf16* __restrict__ out)
{
    const int row = blockIdx.x;
    const int tid = threadIdx.x;
    __shared__ float sh[64];
    const float v = y[(size_t)row * Sc + tid];
    sh[tid] = v;
    __syncthreads();
    for (int o = 32; o > 0; o >>= 1) { if (tid < o) sh[tid] += sh[tid + o]; __syncthreads(); }
    const float mean = sh[0] * (1.f / 64.f);
    __syncthreads();
    const float d = v - mean;
    sh[tid] = d * d;
    __syncthreads();
    for (int o = 32; o > 0; o >>= 1) { if (tid < o) sh[tid] += sh[tid + o]; __syncthreads(); }
    const float var = sh[0] * (1.f / 64.f);
    out[(size_t)row * Sc + tid] = (bf16)(d * rsqrtf(var + 1e-5f));
}

// ymix = (s2i_out + D_param*x_main) * sigmoid(gate half of xz), written as bf16
__global__ void mix_gate(const float* __restrict__ s2i_out, const float* __restrict__ xm_f,
                         const float* __restrict__ Dp, const float* __restrict__ xz,
                         bf16* __restrict__ ymix_bf)
{
    const size_t n = (size_t)Mtok * INNERc;
    size_t idx = (size_t)blockIdx.x * blockDim.x + threadIdx.x;
    if (idx >= n) return;
    const int c   = (int)(idx & (INNERc - 1));
    const int row = (int)(idx >> 11);
    const float g = xz[(size_t)row * (2 * INNERc) + INNERc + c];
    const float val = (s2i_out[idx] + Dp[c] * xm_f[idx]) * (1.f / (1.f + __expf(-g)));
    ymix_bf[idx] = (bf16)val;
}

__global__ void add_residual(const float* __restrict__ a, const float* __restrict__ b,
                             float* __restrict__ out, size_t n)
{
    size_t i = (size_t)blockIdx.x * blockDim.x + threadIdx.x;
    if (i < n) out[i] = a[i] + b[i];
}

__global__ void copy_f32(const float* __restrict__ in, float* __restrict__ out, size_t n)
{
    size_t i = (size_t)blockIdx.x * blockDim.x + threadIdx.x;
    if (i < n) out[i] = in[i];
}

// top-2 gating: one thread per token
__global__ void moe_gating(const bf16* __restrict__ h2, const float* __restrict__ gw,
                           const float* __restrict__ gb, float* __restrict__ wfull)
{
    const int m = blockIdx.x * blockDim.x + threadIdx.x;
    if (m >= Mtok) return;
    float acc[Ec] = {gb[0], gb[1], gb[2], gb[3]};
    const bf16* hr = h2 + (size_t)m * Dc;
    for (int d = 0; d < Dc; d++) {
        const float h = (float)hr[d];
        acc[0] += h * gw[d * Ec + 0];
        acc[1] += h * gw[d * Ec + 1];
        acc[2] += h * gw[d * Ec + 2];
        acc[3] += h * gw[d * Ec + 3];
    }
    int i0 = 0;
    for (int e = 1; e < Ec; e++) if (acc[e] > acc[i0]) i0 = e;
    int i1 = -1;
    for (int e = 0; e < Ec; e++) {
        if (e == i0) continue;
        if (i1 < 0 || acc[e] > acc[i1]) i1 = e;
    }
    const float ea = 1.f;                       // exp(top0 - top0)
    const float eb = __expf(acc[i1] - acc[i0]);
    const float inv = 1.f / (ea + eb);
    for (int e = 0; e < Ec; e++) wfull[(size_t)m * Ec + e] = 0.f;
    wfull[(size_t)m * Ec + i0] = ea * inv;
    wfull[(size_t)m * Ec + i1] = eb * inv;
}

__global__ void gelu_to_bf16(const float* __restrict__ in, bf16* __restrict__ out,
                             size_t n)
{
    size_t i = (size_t)blockIdx.x * blockDim.x + threadIdx.x;
    const size_t stride = (size_t)gridDim.x * blockDim.x;
    for (; i < n; i += stride) {
        const float x = in[i];
        out[i] = (bf16)(0.5f * x * (1.f + erff(x * 0.70710678118654752f)));
    }
}

__global__ void moe_accum(const float* __restrict__ eo, const float* __restrict__ wfull,
                          int e, float* __restrict__ out)
{
    const size_t n = (size_t)Mtok * Dc;
    size_t i = (size_t)blockIdx.x * blockDim.x + threadIdx.x;
    if (i >= n) return;
    const size_t row = i >> 10;                // Dc == 1024
    out[i] += wfull[row * Ec + e] * eo[i];
}

// ---------------------------------------------------------------------------
// Orchestration
// ---------------------------------------------------------------------------

static inline size_t align256(size_t x) { return (x + 255) & ~((size_t)255); }

extern "C" void kernel_launch(void* const* d_in, const int* in_sizes, int n_in,
                              void* d_out, int out_size, void* d_ws, size_t ws_size,
                              hipStream_t stream)
{
    (void)in_sizes; (void)n_in; (void)out_size; (void)ws_size;

    const float* x        = (const float*)d_in[0];
    const float* norm1_w  = (const float*)d_in[1];
    const float* norm2_w  = (const float*)d_in[2];
    const float* in_proj_w= (const float*)d_in[3];
    const float* in_proj_b= (const float*)d_in[4];
    const float* conv_w   = (const float*)d_in[5];
    const float* conv_b   = (const float*)d_in[6];
    const float* dt_w     = (const float*)d_in[7];
    const float* dt_b     = (const float*)d_in[8];
    const float* bp_w     = (const float*)d_in[9];
    const float* bp_b     = (const float*)d_in[10];
    const float* cp_w     = (const float*)d_in[11];
    const float* cp_b     = (const float*)d_in[12];
    const float* s2i_w    = (const float*)d_in[13];
    const float* s2i_b    = (const float*)d_in[14];
    const float* D_param  = (const float*)d_in[15];
    const float* out_w    = (const float*)d_in[16];
    const float* out_b    = (const float*)d_in[17];
    const float* gate_w   = (const float*)d_in[18];
    const float* gate_b   = (const float*)d_in[19];
    const float* e_w1     = (const float*)d_in[20];
    const float* e_b1     = (const float*)d_in[21];
    const float* e_w2     = (const float*)d_in[22];
    const float* e_b2     = (const float*)d_in[23];
    float* out = (float*)d_out;

    // --- workspace carve-up ---
    char* ws = (char*)d_ws;
    size_t off = 0;
    auto carve = [&](size_t bytes) -> void* {
        void* p = ws + off;
        off = align256(off + bytes);
        return p;
    };

    bf16* h1_bf    = (bf16*)carve((size_t)Mtok * Dc * 2);
    bf16* ipw_bf   = (bf16*)carve((size_t)Dc * 2 * INNERc * 2);
    bf16* dtw_bf   = (bf16*)carve((size_t)INNERc * Sc * 2);
    bf16* bpw_bf   = (bf16*)carve((size_t)INNERc * Sc * 2);
    bf16* cpw_bf   = (bf16*)carve((size_t)INNERc * Sc * 2);
    bf16* s2iw_bf  = (bf16*)carve((size_t)Sc * INNERc * 2);
    bf16* outw_bf  = (bf16*)carve((size_t)INNERc * Dc * 2);
    bf16* ew1_bf   = (bf16*)carve((size_t)Ec * Dc * HIDc * 2);
    bf16* ew2_bf   = (bf16*)carve((size_t)Ec * HIDc * Dc * 2);
    float* xz_f    = (float*)carve((size_t)Mtok * 2 * INNERc * 4);
    float* xm_f    = (float*)carve((size_t)Mtok * INNERc * 4);
    bf16*  xm_bf   = (bf16*) carve((size_t)Mtok * INNERc * 2);
    float* dtlin_f = (float*)carve((size_t)Mtok * Sc * 4);
    float* bm_f    = (float*)carve((size_t)Mtok * Sc * 4);
    float* cm_f    = (float*)carve((size_t)Mtok * Sc * 4);
    float* y_f     = (float*)carve((size_t)Mtok * Sc * 4);
    bf16*  y_bf    = (bf16*) carve((size_t)Mtok * Sc * 2);
    float* s2io_f  = (float*)carve((size_t)Mtok * INNERc * 4);
    bf16*  ymix_bf = (bf16*) carve((size_t)Mtok * INNERc * 2);
    float* mixo_f  = (float*)carve((size_t)Mtok * Dc * 4);
    float* x1_f    = (float*)carve((size_t)Mtok * Dc * 4);
    bf16*  h2_bf   = (bf16*) carve((size_t)Mtok * Dc * 2);
    float* wfull_f = (float*)carve((size_t)Mtok * Ec * 4);
    float* hid_f   = (float*)carve((size_t)Mtok * HIDc * 4);
    bf16*  hid_bf  = (bf16*) carve((size_t)Mtok * HIDc * 2);
    float* eo_f    = (float*)carve((size_t)Mtok * Dc * 4);

    // --- weight conversions to bf16 (grid-stride) ---
    const dim3 cvt_grid(2048), cvt_blk(256);
    f32_to_bf16<<<cvt_grid, cvt_blk, 0, stream>>>(in_proj_w, ipw_bf, (size_t)Dc * 2 * INNERc);
    f32_to_bf16<<<cvt_grid, cvt_blk, 0, stream>>>(dt_w, dtw_bf, (size_t)INNERc * Sc);
    f32_to_bf16<<<cvt_grid, cvt_blk, 0, stream>>>(bp_w, bpw_bf, (size_t)INNERc * Sc);
    f32_to_bf16<<<cvt_grid, cvt_blk, 0, stream>>>(cp_w, cpw_bf, (size_t)INNERc * Sc);
    f32_to_bf16<<<cvt_grid, cvt_blk, 0, stream>>>(s2i_w, s2iw_bf, (size_t)Sc * INNERc);
    f32_to_bf16<<<cvt_grid, cvt_blk, 0, stream>>>(out_w, outw_bf, (size_t)INNERc * Dc);
    f32_to_bf16<<<cvt_grid, cvt_blk, 0, stream>>>(e_w1, ew1_bf, (size_t)Ec * Dc * HIDc);
    f32_to_bf16<<<cvt_grid, cvt_blk, 0, stream>>>(e_w2, ew2_bf, (size_t)Ec * HIDc * Dc);

    // --- mixer ---
    rmsnorm_to_bf16<<<Mtok, 256, 0, stream>>>(x, norm1_w, h1_bf, Dc);

    gemm_bf16_wmma<<<dim3((2 * INNERc) / BN, Mtok / BM), 256, 0, stream>>>(
        h1_bf, ipw_bf, in_proj_b, xz_f, 2 * INNERc, Dc);

    {
        const size_t n = (size_t)Mtok * INNERc;
        conv_silu<<<(unsigned)((n + 255) / 256), 256, 0, stream>>>(
            xz_f, conv_w, conv_b, xm_f, xm_bf);
    }

    gemm_bf16_wmma<<<dim3(Sc / BN, Mtok / BM), 256, 0, stream>>>(
        xm_bf, dtw_bf, dt_b, dtlin_f, Sc, INNERc);
    gemm_bf16_wmma<<<dim3(Sc / BN, Mtok / BM), 256, 0, stream>>>(
        xm_bf, bpw_bf, bp_b, bm_f, Sc, INNERc);
    gemm_bf16_wmma<<<dim3(Sc / BN, Mtok / BM), 256, 0, stream>>>(
        xm_bf, cpw_bf, cp_b, cm_f, Sc, INNERc);

    ssm_scan<<<1, Bc * Sc, 0, stream>>>(dtlin_f, bm_f, cm_f, y_f);
    layernorm64_to_bf16<<<Mtok, 64, 0, stream>>>(y_f, y_bf);

    gemm_bf16_wmma<<<dim3(INNERc / BN, Mtok / BM), 256, 0, stream>>>(
        y_bf, s2iw_bf, s2i_b, s2io_f, INNERc, Sc);

    {
        const size_t n = (size_t)Mtok * INNERc;
        mix_gate<<<(unsigned)((n + 255) / 256), 256, 0, stream>>>(
            s2io_f, xm_f, D_param, xz_f, ymix_bf);
    }

    gemm_bf16_wmma<<<dim3(Dc / BN, Mtok / BM), 256, 0, stream>>>(
        ymix_bf, outw_bf, out_b, mixo_f, Dc, INNERc);

    {
        const size_t n = (size_t)Mtok * Dc;
        add_residual<<<(unsigned)((n + 255) / 256), 256, 0, stream>>>(x, mixo_f, x1_f, n);
    }

    // --- MoE ---
    rmsnorm_to_bf16<<<Mtok, 256, 0, stream>>>(x1_f, norm2_w, h2_bf, Dc);
    moe_gating<<<(Mtok + 255) / 256, 256, 0, stream>>>(h2_bf, gate_w, gate_b, wfull_f);

    {
        const size_t n = (size_t)Mtok * Dc;
        copy_f32<<<(unsigned)((n + 255) / 256), 256, 0, stream>>>(x1_f, out, n);
    }

    for (int e = 0; e < Ec; e++) {
        const bf16* w1e = ew1_bf + (size_t)e * Dc * HIDc;
        const bf16* w2e = ew2_bf + (size_t)e * HIDc * Dc;
        const float* b1e = e_b1 + (size_t)e * HIDc;
        const float* b2e = e_b2 + (size_t)e * Dc;

        gemm_bf16_wmma<<<dim3(HIDc / BN, Mtok / BM), 256, 0, stream>>>(
            h2_bf, w1e, b1e, hid_f, HIDc, Dc);

        gelu_to_bf16<<<cvt_grid, cvt_blk, 0, stream>>>(hid_f, hid_bf, (size_t)Mtok * HIDc);

        gemm_bf16_wmma<<<dim3(Dc / BN, Mtok / BM), 256, 0, stream>>>(
            hid_bf, w2e, b2e, eo_f, Dc, HIDc);

        const size_t n = (size_t)Mtok * Dc;
        moe_accum<<<(unsigned)((n + 255) / 256), 256, 0, stream>>>(eo_f, wfull_f, e, out);
    }
}
